// MPGSA_50697793962809
// MI455X (gfx1250) — matrix operand order repaired and checked by reading, not compiled, not run on hardware
//
#include <hip/hip_runtime.h>
#include <hip/hip_bf16.h>
#include <math.h>

// ---------------- types for CDNA5 WMMA ----------------
typedef __bf16 bf16;
typedef __attribute__((ext_vector_type(16))) __bf16 v16bf;
typedef __attribute__((ext_vector_type(8)))  float  v8f;

// ---------------- problem constants ----------------
static constexpr int Bn = 4, Hn = 128, Wn = 128, DIMn = 64;
static constexpr int HEADSn = 8, DRn = 16;
static constexpr int HWn = Hn * Wn;          // 16384
static constexpr int ROWSn = Bn * HWn;       // 65536
static constexpr int HID3n = 48, HIDXn = 96;

// ---------------- workspace layout (bytes) ----------------
static constexpr size_t OFF_WCOMB_F32 = 0;                                   // 64*48 f32
static constexpr size_t OFF_WCOMB_BF  = OFF_WCOMB_F32 + 64 * 48 * 4;         // 64*48 bf16
static constexpr size_t OFF_QDIM      = 18432;                               // [B][HW][16] f32
static constexpr size_t OFF_KDIM      = OFF_QDIM   + (size_t)ROWSn * 16 * 4;
static constexpr size_t OFF_VDIM      = OFF_KDIM   + (size_t)ROWSn * 16 * 4;
static constexpr size_t OFF_QMIX      = OFF_VDIM   + (size_t)ROWSn * 16 * 4;
static constexpr size_t OFF_KMIX      = OFF_QMIX   + (size_t)ROWSn * 16 * 4;
static constexpr size_t OFF_VMIX      = OFF_KMIX   + (size_t)ROWSn * 16 * 4;
static constexpr size_t NFREQ         = (size_t)Bn * 16 * 128 * 65;          // 532480
static constexpr size_t OFF_ROWFFT    = OFF_VMIX   + (size_t)ROWSn * 16 * 4; // float2
static constexpr size_t OFF_AMP       = OFF_ROWFFT + NFREQ * 8;
static constexpr size_t OFF_PHASE     = OFF_AMP    + NFREQ * 4;
static constexpr size_t OFF_ALO       = OFF_PHASE  + NFREQ * 4;
static constexpr size_t OFF_AHI       = OFF_ALO    + NFREQ * 4;
static constexpr size_t OFF_PHOUT     = OFF_AHI    + NFREQ * 4;
static constexpr size_t OFF_Y         = OFF_PHOUT  + NFREQ * 4;              // float2
static constexpr size_t OFF_COLIFFT   = OFF_Y      + NFREQ * 8;              // float2
static constexpr size_t OFF_LL        = OFF_COLIFFT+ NFREQ * 8;              // [B][64][64][16]
static constexpr size_t OFF_HIGH      = OFF_LL     + (size_t)Bn * 4096 * 16 * 4;
static constexpr size_t OFF_M1        = OFF_HIGH   + (size_t)Bn * 4096 * 48 * 4;
static constexpr size_t OFF_M2        = OFF_M1     + (size_t)Bn * 4096 * 96 * 4;
static constexpr size_t OFF_GRAM      = OFF_M2     + (size_t)Bn * 4096 * 96 * 4; // [B][3][256] f32
static constexpr size_t OFF_PWF       = OFF_GRAM   + (size_t)Bn * 3 * 256 * 4;   // [B][16][64] f32
static constexpr size_t OFF_PWBF      = OFF_PWF    + (size_t)Bn * 16 * 64 * 4;   // [B][32][64] bf16

// ---------------- helpers ----------------
__device__ __forceinline__ bf16 f2bf(float f) {
  unsigned u = __float_as_uint(f);
  u += 0x7fffu + ((u >> 16) & 1u);
  unsigned short h = (unsigned short)(u >> 16);
  return __builtin_bit_cast(__bf16, h);
}
__device__ __forceinline__ float geluf(float x) {
  return 0.5f * x * (1.0f + erff(x * 0.70710678118654752f));
}

// ---------------- K0: Wcomb = [Wq@Wr | Wk@Wr | Wv@Wr]  (64 x 48) ----------------
__global__ __launch_bounds__(256) void k_wcomb(const float* __restrict__ Wq,
    const float* __restrict__ Wk, const float* __restrict__ Wv,
    const float* __restrict__ Wr, float* __restrict__ wf, bf16* __restrict__ wb) {
  int t = blockIdx.x * 256 + threadIdx.x;
  if (t >= 64 * 48) return;
  int i = t / 48, j = t % 48;
  int which = j >> 4, jj = j & 15;
  const float* W = (which == 0) ? Wq : ((which == 1) ? Wk : Wv);
  float acc = 0.f;
  for (int m = 0; m < 512; ++m) acc += W[i * 512 + m] * Wr[m * 16 + jj];
  wf[t] = acc;
  wb[t] = f2bf(acc);
}

__global__ __launch_bounds__(256) void k_zero(float* __restrict__ p, int n) {
  int t = blockIdx.x * 256 + threadIdx.x;
  if (t < n) p[t] = 0.f;
}

// ---------------- K1: qkv_dim = x @ Wcomb  (WMMA bf16) ----------------
__global__ __launch_bounds__(256) void k_qkv_dim(const float* __restrict__ x,
    const bf16* __restrict__ Wc, float* __restrict__ qd,
    float* __restrict__ kd, float* __restrict__ vd) {
  int wave = threadIdx.x >> 5, lane = threadIdx.x & 31;
  int tile = blockIdx.x * 8 + wave;        // 4096 tiles of 16 rows
  int r0 = tile * 16;
  int m = lane & 15, g = lane >> 4, cb = g * 8, kb = g * 16;
  const float* xr = x + (size_t)(r0 + m) * 64;
  v16bf a0, a1;
#pragma unroll
  for (int j = 0; j < 8; ++j) {
    a0[j]     = f2bf(xr[cb + j]);
    a0[j + 8] = f2bf(xr[16 + cb + j]);
    a1[j]     = f2bf(xr[32 + cb + j]);
    a1[j + 8] = f2bf(xr[48 + cb + j]);
  }
  float* outs[3] = {qd, kd, vd};
#pragma unroll
  for (int nt = 0; nt < 3; ++nt) {
    int n = nt * 16 + m;
    v16bf b0, b1;
#pragma unroll
    for (int j = 0; j < 16; ++j) {
      b0[j] = Wc[(cb + j + (j & 8)) * 48 + n];
      b1[j] = Wc[(32 + cb + j + (j & 8)) * 48 + n];
    }
    v8f c = {};
    c = __builtin_amdgcn_wmma_f32_16x16x32_bf16(false, a0, false, b0, (short)0, c, false, false);
    c = __builtin_amdgcn_wmma_f32_16x16x32_bf16(false, a1, false, b1, (short)0, c, false, false);
    float* o = outs[nt];
#pragma unroll
    for (int i = 0; i < 8; ++i)
      o[(size_t)(r0 + i + 8 * g) * 16 + m] = c[i];
  }
}

// ---------------- K2: SpatialUnit on q_dim ----------------
__global__ __launch_bounds__(256) void k_spatial(const float* __restrict__ qd,
    const float* __restrict__ dw, const float* __restrict__ pw,
    const float* __restrict__ gamma, float* __restrict__ qmix) {
  int t = blockIdx.x * 256 + threadIdx.x;  // b*16384 + y*128 + x
  int b = t >> 14, yx = t & 16383, y = yx >> 7, x = yx & 127;
  const float* base = qd + (size_t)b * HWn * 16;
  float g9[16];
#pragma unroll
  for (int c = 0; c < 16; ++c) g9[c] = 0.f;
  for (int dy = -1; dy <= 1; ++dy) {
    int yy = y + dy; if ((unsigned)yy >= 128u) continue;
    for (int dx = -1; dx <= 1; ++dx) {
      int xx = x + dx; if ((unsigned)xx >= 128u) continue;
      const float* p = base + (size_t)(yy * 128 + xx) * 16;
      int wi = (dy + 1) * 3 + (dx + 1);
#pragma unroll
      for (int c = 0; c < 16; ++c) g9[c] += dw[c * 9 + wi] * p[c];
    }
  }
  float ge[16];
#pragma unroll
  for (int c = 0; c < 16; ++c) ge[c] = geluf(g9[c]);
  const float* self = base + (size_t)yx * 16;
  float* o = qmix + (size_t)t * 16;
#pragma unroll
  for (int c = 0; c < 16; ++c) {
    float z = 0.f;
#pragma unroll
    for (int ci = 0; ci < 16; ++ci) z += pw[c * 16 + ci] * ge[ci];
    o[c] = self[c] + gamma[c] * z;
  }
}

// ---------------- K3: Haar DWT + BN + mlp_in (torch interleaved chunk layout) ----------------
__global__ __launch_bounds__(256) void k_wave1(const float* __restrict__ vd,
    const float* __restrict__ bn_g, const float* __restrict__ bn_b,
    const float* __restrict__ mlp_in, float* __restrict__ LL,
    float* __restrict__ high, float* __restrict__ m1) {
  int t = blockIdx.x * 256 + threadIdx.x;  // b*4096 + i*64 + j
  int b = t >> 12, ij = t & 4095, i = ij >> 6, j = ij & 63;
  const float* base = vd + (size_t)b * HWn * 16;
  float y[64];
#pragma unroll
  for (int c = 0; c < 16; ++c) {
    float s00 = base[((2 * i) * 128 + 2 * j) * 16 + c];
    float s01 = base[((2 * i) * 128 + 2 * j + 1) * 16 + c];
    float s10 = base[((2 * i + 1) * 128 + 2 * j) * 16 + c];
    float s11 = base[((2 * i + 1) * 128 + 2 * j + 1) * 16 + c];
    y[c * 4 + 0] = 0.5f * (s00 + s01 + s10 + s11);
    y[c * 4 + 1] = 0.5f * (s00 - s01 + s10 - s11);
    y[c * 4 + 2] = 0.5f * (s00 + s01 - s10 - s11);
    y[c * 4 + 3] = 0.5f * (s00 - s01 - s10 + s11);
  }
  float* LLp = LL + (size_t)t * 16;
#pragma unroll
  for (int c = 0; c < 16; ++c) LLp[c] = y[c];
  const float bnscale = rsqrtf(1.0f + 1e-5f);
  float n[48];
  float* hp = high + (size_t)t * 48;
#pragma unroll
  for (int c = 0; c < 48; ++c) {
    float hv = y[16 + c];
    hp[c] = hv;
    n[c] = hv * (bn_g[c] * bnscale) + bn_b[c];
  }
  float* mp = m1 + (size_t)t * 96;
  for (int o = 0; o < 96; ++o) {
    float acc = 0.f;
#pragma unroll
    for (int c = 0; c < 48; ++c) acc += mlp_in[o * 48 + c] * n[c];
    mp[o] = acc;
  }
}

// ---------------- K4: depthwise 3x3 (96ch, 64x64) + GELU ----------------
__global__ __launch_bounds__(256) void k_wave2(const float* __restrict__ m1,
    const float* __restrict__ mdw, float* __restrict__ m2) {
  int t = blockIdx.x * 256 + threadIdx.x;  // (b*4096+ij)*12 + og
  int og = t % 12, pix = t / 12;
  int b = pix >> 12, ij = pix & 4095, i = ij >> 6, j = ij & 63;
  int o0 = og * 8;
  const float* base = m1 + (size_t)b * 4096 * 96;
  float acc[8] = {0.f, 0.f, 0.f, 0.f, 0.f, 0.f, 0.f, 0.f};
  for (int dy = -1; dy <= 1; ++dy) {
    int yy = i + dy; if ((unsigned)yy >= 64u) continue;
    for (int dx = -1; dx <= 1; ++dx) {
      int xx = j + dx; if ((unsigned)xx >= 64u) continue;
      const float* p = base + (size_t)(yy * 64 + xx) * 96 + o0;
      int wi = (dy + 1) * 3 + (dx + 1);
#pragma unroll
      for (int oo = 0; oo < 8; ++oo) acc[oo] += mdw[(o0 + oo) * 9 + wi] * p[oo];
    }
  }
  float* mp = m2 + (size_t)pix * 96 + o0;
#pragma unroll
  for (int oo = 0; oo < 8; ++oo) mp[oo] = geluf(acc[oo]);
}

// ---------------- K5: mlp_out + residual + IWT scatter ----------------
__global__ __launch_bounds__(256) void k_wave3(const float* __restrict__ high,
    const float* __restrict__ m2, const float* __restrict__ mlp_out,
    const float* __restrict__ res_scale, const float* __restrict__ LL,
    float* __restrict__ vmix) {
  int t = blockIdx.x * 256 + threadIdx.x;
  int b = t >> 12, ij = t & 4095, i = ij >> 6, j = ij & 63;
  const float* hp = high + (size_t)t * 48;
  const float* m2p = m2 + (size_t)t * 96;
  const float* LLp = LL + (size_t)t * 16;
  float rs = res_scale[0];
  float he[48];
  for (int c = 0; c < 48; ++c) {
    float acc = 0.f;
#pragma unroll
    for (int o = 0; o < 96; ++o) acc += mlp_out[c * 96 + o] * m2p[o];
    he[c] = hp[c] + rs * acc;
  }
  float* vb = vmix + (size_t)b * HWn * 16;
  int r0 = 2 * i, c0 = 2 * j;
#pragma unroll
  for (int c = 0; c < 16; ++c) {
    vb[((size_t)r0 * 128 + c0) * 16 + c]           = LLp[c];
    vb[((size_t)r0 * 128 + c0 + 1) * 16 + c]       = he[c];
    vb[((size_t)(r0 + 1) * 128 + c0) * 16 + c]     = he[16 + c];
    vb[((size_t)(r0 + 1) * 128 + c0 + 1) * 16 + c] = he[32 + c];
  }
}

// ---------------- K6: forward row DFT (real -> 65 bins) ----------------
__global__ __launch_bounds__(128) void k_fft_row(const float* __restrict__ kd,
                                                 float2* __restrict__ rowfft) {
  __shared__ float row[128];
  __shared__ float2 tw[128];
  int blk = blockIdx.x;               // (b*16+c)*128 + r
  int bc = blk >> 7, r = blk & 127;
  int b = bc >> 4, c = bc & 15;
  int t = threadIdx.x;
  row[t] = kd[((size_t)b * HWn + r * 128 + t) * 16 + c];
  float s, co;
  sincosf(-6.28318530717958647692f * (float)t / 128.0f, &s, &co);
  tw[t] = make_float2(co, s);
  __syncthreads();
  if (t < 65) {
    float re = 0.f, im = 0.f;
    for (int n = 0; n < 128; ++n) {
      float2 w = tw[(t * n) & 127];
      re += row[n] * w.x;
      im += row[n] * w.y;
    }
    rowfft[(size_t)bc * 128 * 65 + (size_t)r * 65 + t] = make_float2(re, im);
  }
}

// ---------------- K7: forward column DFT + ortho + amp/phase ----------------
__global__ __launch_bounds__(128) void k_fft_col(const float2* __restrict__ rowfft,
    float* __restrict__ amp, float* __restrict__ phase) {
  __shared__ float zr[128], zi[128];
  __shared__ float2 tw[128];
  int blk = blockIdx.x;               // bc*65 + col
  int bc = blk / 65, col = blk % 65;
  int t = threadIdx.x;
  float2 z = rowfft[(size_t)bc * 128 * 65 + (size_t)t * 65 + col];
  zr[t] = z.x; zi[t] = z.y;
  float s, co;
  sincosf(-6.28318530717958647692f * (float)t / 128.0f, &s, &co);
  tw[t] = make_float2(co, s);
  __syncthreads();
  float re = 0.f, im = 0.f;
  for (int n = 0; n < 128; ++n) {
    float2 w = tw[(t * n) & 127];
    re += zr[n] * w.x - zi[n] * w.y;
    im += zr[n] * w.y + zi[n] * w.x;
  }
  re *= (1.0f / 128.0f);  // ortho for 128x128
  im *= (1.0f / 128.0f);
  size_t idx = (size_t)bc * 128 * 65 + (size_t)t * 65 + col;
  amp[idx] = sqrtf(re * re + im * im);
  phase[idx] = atan2f(im, re);
}

// ---------------- K8: freq-domain pointwise convs + masks + GELU ----------------
__global__ __launch_bounds__(256) void k_freq_pw(const float* __restrict__ amp,
    const float* __restrict__ phase, const float* __restrict__ al_pw,
    const float* __restrict__ ah_pw, const float* __restrict__ ph_pw,
    const float* __restrict__ ph_scale_p, float* __restrict__ alo,
    float* __restrict__ ahi, float* __restrict__ phout) {
  int t = blockIdx.x * 256 + threadIdx.x;  // b*(128*65) + u*65 + v
  int b = t / (128 * 65), uv = t % (128 * 65), u = uv / 65, v = uv % 65;
  float yy = (float)u * (1.0f / 127.0f), xx = (float)v * (1.0f / 64.0f);
  float rr = sqrtf(yy * yy + xx * xx);
  float low = 1.0f / (1.0f + __expf(-(0.25f - rr) * 20.0f));
  float hi = 1.0f - low;
  float aL[16], aH[16], ph[16];
#pragma unroll
  for (int ci = 0; ci < 16; ++ci) {
    size_t idx = ((size_t)(b * 16 + ci) * 128 + u) * 65 + v;
    float a = amp[idx];
    aL[ci] = a * low; aH[ci] = a * hi; ph[ci] = phase[idx];
  }
  float phs = ph_scale_p[0];
#pragma unroll
  for (int c = 0; c < 16; ++c) {
    float sl = 0.f, sh = 0.f, sp = 0.f;
#pragma unroll
    for (int ci = 0; ci < 16; ++ci) {
      sl += al_pw[c * 16 + ci] * aL[ci];
      sh += ah_pw[c * 16 + ci] * aH[ci];
      sp += ph_pw[c * 16 + ci] * ph[ci];
    }
    size_t idx = ((size_t)(b * 16 + c) * 128 + u) * 65 + v;
    alo[idx] = geluf(sl);
    ahi[idx] = geluf(sh);
    phout[idx] = ph[c] + phs * sp;
  }
}

// ---------------- K9: freq depthwise 3x3 + recombine to complex ----------------
__global__ __launch_bounds__(256) void k_freq_dw(const float* __restrict__ alo,
    const float* __restrict__ ahi, const float* __restrict__ phout,
    const float* __restrict__ al_dw, const float* __restrict__ ah_dw,
    float2* __restrict__ Y) {
  int t = blockIdx.x * 256 + threadIdx.x;  // (b*16+c)*(128*65) + u*65 + v
  int bc = t / (128 * 65), uv = t % (128 * 65), u = uv / 65, v = uv % 65;
  int c = bc & 15;
  const float* lb = alo + (size_t)bc * 128 * 65;
  const float* hb = ahi + (size_t)bc * 128 * 65;
  float acc = 0.f;
  for (int dy = -1; dy <= 1; ++dy) {
    int uu = u + dy; if ((unsigned)uu >= 128u) continue;
    for (int dx = -1; dx <= 1; ++dx) {
      int vv = v + dx; if ((unsigned)vv >= 65u) continue;
      int wi = (dy + 1) * 3 + (dx + 1);
      size_t o = (size_t)uu * 65 + vv;
      acc += al_dw[c * 9 + wi] * lb[o] + ah_dw[c * 9 + wi] * hb[o];
    }
  }
  float p = phout[t];
  float s, co;
  sincosf(p, &s, &co);
  Y[t] = make_float2(acc * co, acc * s);
}

// ---------------- K10: inverse column DFT (complex) ----------------
__global__ __launch_bounds__(128) void k_ifft_col(const float2* __restrict__ Y,
                                                  float2* __restrict__ colifft) {
  __shared__ float yr[128], yi[128];
  __shared__ float2 tw[128];
  int blk = blockIdx.x;               // bc*65 + col
  int bc = blk / 65, col = blk % 65;
  int t = threadIdx.x;
  float2 z = Y[(size_t)bc * 128 * 65 + (size_t)t * 65 + col];
  yr[t] = z.x; yi[t] = z.y;
  float s, co;
  sincosf(6.28318530717958647692f * (float)t / 128.0f, &s, &co);
  tw[t] = make_float2(co, s);
  __syncthreads();
  float re = 0.f, im = 0.f;
  for (int k = 0; k < 128; ++k) {
    float2 w = tw[(t * k) & 127];
    re += yr[k] * w.x - yi[k] * w.y;
    im += yr[k] * w.y + yi[k] * w.x;
  }
  colifft[(size_t)bc * 128 * 65 + (size_t)t * 65 + col] = make_float2(re, im);
}

// ---------------- K11: inverse row real DFT (Hermitian) ----------------
__global__ __launch_bounds__(128) void k_ifft_row(const float2* __restrict__ colifft,
                                                  float* __restrict__ kmix) {
  __shared__ float wr[65], wi[65];
  __shared__ float2 tw[128];
  int blk = blockIdx.x;               // bc*128 + r
  int bc = blk >> 7, r = blk & 127;
  int b = bc >> 4, c = bc & 15;
  int t = threadIdx.x;
  if (t < 65) {
    float2 z = colifft[(size_t)bc * 128 * 65 + (size_t)r * 65 + t];
    wr[t] = z.x; wi[t] = z.y;
  }
  float s, co;
  sincosf(6.28318530717958647692f * (float)t / 128.0f, &s, &co);
  tw[t] = make_float2(co, s);
  __syncthreads();
  float acc = wr[0] + (((t & 1) != 0) ? -wr[64] : wr[64]);
  for (int k = 1; k < 64; ++k) {
    float2 w = tw[(k * t) & 127];
    acc += 2.0f * (wr[k] * w.x - wi[k] * w.y);
  }
  acc *= (1.0f / 128.0f);  // ortho inverse
  kmix[((size_t)b * HWn + r * 128 + t) * 16 + c] = acc;
}

// ---------------- K12: Gram matrices kq/kk/qq per batch (WMMA bf16 + f32 atomics) ----
__global__ __launch_bounds__(256) void k_gram(const float* __restrict__ qmix,
    const float* __restrict__ kmix, float* __restrict__ gram) {
  int b = blockIdx.x >> 4;            // 16 blocks per batch
  int blk = blockIdx.x & 15;
  int wave = threadIdx.x >> 5, lane = threadIdx.x & 31;
  int m = lane & 15, g = lane >> 4, cb = g * 8, kb = g * 16;
  const float* qm = qmix + (size_t)b * HWn * 16;
  const float* km = kmix + (size_t)b * HWn * 16;
  v8f ckq = {}, ckk = {}, cqq = {};
  int wid = blk * 8 + wave;           // 0..127 ; 4 chunks of 32 each
  for (int ch = 0; ch < 4; ++ch) {
    int n0 = (wid * 4 + ch) * 32;
    __builtin_prefetch(km + (size_t)(n0 + 32) * 16, 0, 1);
    __builtin_prefetch(qm + (size_t)(n0 + 32) * 16, 0, 1);
    v16bf ak, aq, bk, bq;
#pragma unroll
    for (int j = 0; j < 8; ++j) {
      ak[j]     = f2bf(km[(size_t)(n0 + cb + j) * 16 + m]);
      ak[j + 8] = f2bf(km[(size_t)(n0 + 16 + cb + j) * 16 + m]);
      aq[j]     = f2bf(qm[(size_t)(n0 + cb + j) * 16 + m]);
      aq[j + 8] = f2bf(qm[(size_t)(n0 + 16 + cb + j) * 16 + m]);
    }
#pragma unroll
    for (int j = 0; j < 16; ++j) {
      bk[j] = f2bf(km[(size_t)(n0 + cb + j + (j & 8)) * 16 + m]);
      bq[j] = f2bf(qm[(size_t)(n0 + cb + j + (j & 8)) * 16 + m]);
    }
    ckq = __builtin_amdgcn_wmma_f32_16x16x32_bf16(false, ak, false, bq, (short)0, ckq, false, false);
    ckk = __builtin_amdgcn_wmma_f32_16x16x32_bf16(false, ak, false, bk, (short)0, ckk, false, false);
    cqq = __builtin_amdgcn_wmma_f32_16x16x32_bf16(false, aq, false, bq, (short)0, cqq, false, false);
  }
  float* gb = gram + (size_t)b * 3 * 256;
#pragma unroll
  for (int i = 0; i < 8; ++i) {
    int row = i + 8 * g, col = m;
    atomicAdd(&gb[row * 16 + col],       ckq[i]);
    atomicAdd(&gb[256 + row * 16 + col], ckk[i]);
    atomicAdd(&gb[512 + row * 16 + col], cqq[i]);
  }
}

// ---------------- K13: per-batch attention fold -> P_b[16][64] ----------------
__global__ __launch_bounds__(256) void k_attn(const float* __restrict__ gram,
    const float* __restrict__ We, const float* __restrict__ Wproj,
    const float* __restrict__ rescale, float* __restrict__ pwf,
    bf16* __restrict__ pwbf) {
  __shared__ float sG[768];
  __shared__ float U[16 * 64];
  __shared__ float Hkq[16 * 64], Hkk[16 * 64], Hqq[16 * 64];
  __shared__ float L[64 * 64];
  __shared__ float nk[64], nq[64];
  __shared__ float Pp[16 * 64];
  __shared__ float PW[16 * 64];
  int b = blockIdx.x, t = threadIdx.x;
  for (int i = t; i < 768; i += 256) sG[i] = gram[b * 768 + i];
  for (int i = t; i < 1024; i += 256) PW[i] = 0.f;
  __syncthreads();
  for (int h = 0; h < HEADSn; ++h) {
    for (int i = t; i < 1024; i += 256) {
      int ii = i >> 6, d = i & 63;
      U[i] = We[ii * 512 + h * 64 + d];
    }
    __syncthreads();
    for (int i = t; i < 1024; i += 256) {
      int ii = i >> 6, d = i & 63;
      float a0 = 0.f, a1 = 0.f, a2 = 0.f;
#pragma unroll
      for (int j = 0; j < 16; ++j) {
        float u = U[j * 64 + d];
        a0 += sG[ii * 16 + j] * u;
        a1 += sG[256 + ii * 16 + j] * u;
        a2 += sG[512 + ii * 16 + j] * u;
      }
      Hkq[i] = a0; Hkk[i] = a1; Hqq[i] = a2;
    }
    __syncthreads();
    if (t < 64) {
      int d = t;
      float s = 0.f, s2 = 0.f;
#pragma unroll
      for (int i2 = 0; i2 < 16; ++i2) {
        s  += U[i2 * 64 + d] * Hkk[i2 * 64 + d];
        s2 += U[i2 * 64 + d] * Hqq[i2 * 64 + d];
      }
      nk[d] = fmaxf(sqrtf(fmaxf(s, 0.f)), 1e-12f);
      nq[d] = fmaxf(sqrtf(fmaxf(s2, 0.f)), 1e-12f);
    }
    __syncthreads();
    float rs = rescale[h];
    for (int i = t; i < 4096; i += 256) {
      int d = i >> 6, e = i & 63;
      float s = 0.f;
#pragma unroll
      for (int i2 = 0; i2 < 16; ++i2) s += U[i2 * 64 + d] * Hkq[i2 * 64 + e];
      L[i] = s / (nk[d] * nq[e]) * rs;
    }
    __syncthreads();
    if (t < 64) {   // softmax over e per row d
      int d = t;
      float mx = -1e30f;
      for (int e = 0; e < 64; ++e) mx = fmaxf(mx, L[d * 64 + e]);
      float s = 0.f;
      for (int e = 0; e < 64; ++e) { float v = __expf(L[d * 64 + e] - mx); L[d * 64 + e] = v; s += v; }
      float inv = 1.0f / s;
      for (int e = 0; e < 64; ++e) L[d * 64 + e] *= inv;
    }
    __syncthreads();
    for (int i = t; i < 1024; i += 256) {
      int ii = i >> 6, d = i & 63;
      float s = 0.f;
      for (int e = 0; e < 64; ++e) s += U[ii * 64 + e] * L[d * 64 + e];
      Pp[i] = s;
    }
    __syncthreads();
    for (int i = t; i < 1024; i += 256) {
      int ii = i >> 6, c = i & 63;
      float s = 0.f;
      for (int d = 0; d < 64; ++d) s += Pp[ii * 64 + d] * Wproj[(h * 64 + d) * 64 + c];
      PW[i] += s;
    }
    __syncthreads();
  }
  for (int i = t; i < 1024; i += 256) pwf[b * 1024 + i] = PW[i];
  for (int i = t; i < 2048; i += 256) {   // [32][64], K zero-padded to 32
    int kk = i >> 6, c = i & 63;
    pwbf[b * 2048 + i] = (kk < 16) ? f2bf(PW[kk * 64 + c]) : f2bf(0.f);
  }
}

// ---------------- K14: out = v_mix @ P_b + bproj  (WMMA bf16, K padded) ----------
__global__ __launch_bounds__(256) void k_out(const float* __restrict__ vmix,
    const bf16* __restrict__ pwbf, const float* __restrict__ bproj,
    float* __restrict__ out) {
  int wave = threadIdx.x >> 5, lane = threadIdx.x & 31;
  int tile = blockIdx.x * 8 + wave;
  int r0 = tile * 16;
  int b = r0 >> 14;
  int m = lane & 15, g = lane >> 4, cb = g * 8, kb = g * 16;
  const float* vr = vmix + (size_t)(r0 + m) * 16;
  v16bf a;
#pragma unroll
  for (int j = 0; j < 8; ++j) {
    a[j] = f2bf(vr[cb + j]);    // K = cb+j in [0,16)
    a[j + 8] = f2bf(0.0f);      // K >= 16 : zero padding
  }
  const bf16* Bp = pwbf + (size_t)b * 2048;
#pragma unroll
  for (int nt = 0; nt < 4; ++nt) {
    int n = nt * 16 + m;
    v16bf bm;
#pragma unroll
    for (int j = 0; j < 16; ++j) bm[j] = Bp[(cb + j + (j & 8)) * 64 + n];
    v8f c = {};
    c = __builtin_amdgcn_wmma_f32_16x16x32_bf16(false, a, false, bm, (short)0, c, false, false);
    float bb = bproj[n];
#pragma unroll
    for (int i = 0; i < 8; ++i)
      out[(size_t)(r0 + i + 8 * g) * 64 + n] = c[i] + bb;
  }
}

// ---------------- host side ----------------
extern "C" void kernel_launch(void* const* d_in, const int* in_sizes, int n_in,
                              void* d_out, int out_size, void* d_ws, size_t ws_size,
                              hipStream_t stream) {
  const float* x        = (const float*)d_in[0];
  const float* Wq       = (const float*)d_in[1];
  const float* Wk       = (const float*)d_in[2];
  const float* Wv       = (const float*)d_in[3];
  const float* Wr       = (const float*)d_in[4];
  const float* We       = (const float*)d_in[5];
  const float* qs_dw    = (const float*)d_in[6];
  const float* qs_pw    = (const float*)d_in[7];
  const float* qs_gamma = (const float*)d_in[8];
  const float* al_pw    = (const float*)d_in[9];
  const float* al_dw    = (const float*)d_in[10];
  const float* ah_pw    = (const float*)d_in[11];
  const float* ah_dw    = (const float*)d_in[12];
  const float* ph_pw    = (const float*)d_in[13];
  const float* ph_scale = (const float*)d_in[14];
  const float* bn_gamma = (const float*)d_in[15];
  const float* bn_beta  = (const float*)d_in[16];
  const float* mlp_in   = (const float*)d_in[17];
  const float* mlp_dw   = (const float*)d_in[18];
  const float* mlp_out  = (const float*)d_in[19];
  const float* res_sc   = (const float*)d_in[20];
  const float* rescale  = (const float*)d_in[21];
  const float* Wproj    = (const float*)d_in[22];
  const float* bproj    = (const float*)d_in[23];

  char* ws = (char*)d_ws;
  float*  wcf    = (float*)(ws + OFF_WCOMB_F32);
  bf16*   wcb    = (bf16*)(ws + OFF_WCOMB_BF);
  float*  qdim   = (float*)(ws + OFF_QDIM);
  float*  kdim   = (float*)(ws + OFF_KDIM);
  float*  vdim   = (float*)(ws + OFF_VDIM);
  float*  qmix   = (float*)(ws + OFF_QMIX);
  float*  kmix   = (float*)(ws + OFF_KMIX);
  float*  vmix   = (float*)(ws + OFF_VMIX);
  float2* rowfft = (float2*)(ws + OFF_ROWFFT);
  float*  amp    = (float*)(ws + OFF_AMP);
  float*  phase  = (float*)(ws + OFF_PHASE);
  float*  alo    = (float*)(ws + OFF_ALO);
  float*  ahi    = (float*)(ws + OFF_AHI);
  float*  phout  = (float*)(ws + OFF_PHOUT);
  float2* Ybuf   = (float2*)(ws + OFF_Y);
  float2* colif  = (float2*)(ws + OFF_COLIFFT);
  float*  LLb    = (float*)(ws + OFF_LL);
  float*  highb  = (float*)(ws + OFF_HIGH);
  float*  m1b    = (float*)(ws + OFF_M1);
  float*  m2b    = (float*)(ws + OFF_M2);
  float*  gram   = (float*)(ws + OFF_GRAM);
  float*  pwf    = (float*)(ws + OFF_PWF);
  bf16*   pwbf   = (bf16*)(ws + OFF_PWBF);

  k_wcomb   <<<dim3(12),   dim3(256), 0, stream>>>(Wq, Wk, Wv, Wr, wcf, wcb);
  k_zero    <<<dim3(12),   dim3(256), 0, stream>>>(gram, Bn * 3 * 256);
  k_qkv_dim <<<dim3(512),  dim3(256), 0, stream>>>(x, wcb, qdim, kdim, vdim);
  k_spatial <<<dim3(256),  dim3(256), 0, stream>>>(qdim, qs_dw, qs_pw, qs_gamma, qmix);
  k_wave1   <<<dim3(64),   dim3(256), 0, stream>>>(vdim, bn_gamma, bn_beta, mlp_in, LLb, highb, m1b);
  k_wave2   <<<dim3(768),  dim3(256), 0, stream>>>(m1b, mlp_dw, m2b);
  k_wave3   <<<dim3(64),   dim3(256), 0, stream>>>(highb, m2b, mlp_out, res_sc, LLb, vmix);
  k_fft_row <<<dim3(8192), dim3(128), 0, stream>>>(kdim, rowfft);
  k_fft_col <<<dim3(4160), dim3(128), 0, stream>>>(rowfft, amp, phase);
  k_freq_pw <<<dim3(130),  dim3(256), 0, stream>>>(amp, phase, al_pw, ah_pw, ph_pw, ph_scale, alo, ahi, phout);
  k_freq_dw <<<dim3(2080), dim3(256), 0, stream>>>(alo, ahi, phout, al_dw, ah_dw, Ybuf);
  k_ifft_col<<<dim3(4160), dim3(128), 0, stream>>>(Ybuf, colif);
  k_ifft_row<<<dim3(8192), dim3(128), 0, stream>>>(colif, kmix);
  k_gram    <<<dim3(64),   dim3(256), 0, stream>>>(qmix, kmix, gram);
  k_attn    <<<dim3(Bn),   dim3(256), 0, stream>>>(gram, We, Wproj, rescale, pwf, pwbf);
  k_out     <<<dim3(512),  dim3(256), 0, stream>>>(vmix, pwbf, bproj, (float*)d_out);
}
